// BatchMultiHeadAttentionUnit_45956150067409
// MI455X (gfx1250) — compile-verified, matrix operand order checked by reading, not run on hardware
//
#include <hip/hip_runtime.h>

#define B_  4
#define S_  2048
#define D_  1024
#define NH_ 16
#define HD_ 64

typedef __bf16 bf16_t;
typedef __attribute__((ext_vector_type(16))) __bf16 v16bf;
typedef __attribute__((ext_vector_type(8)))  __bf16 v8bf;
typedef __attribute__((ext_vector_type(8)))  float  v8f;

static __device__ __forceinline__ v8f vzero8() {
  v8f v;
#pragma unroll
  for (int i = 0; i < 8; ++i) v[i] = 0.0f;
  return v;
}

// Build a 16x32 bf16 A-fragment (or 32x16 B-fragment) for wave32 WMMA.
// Per ISA layout: lanes 0-15 hold K {0..7, 16..23}, lanes 16-31 hold K {8..15, 24..31}.
static __device__ __forceinline__ v16bf load_frag(const bf16_t* p, int lo) {
  v8bf x = *(const v8bf*)(p + lo);
  v8bf y = *(const v8bf*)(p + 16 + lo);
  v16bf f;
#pragma unroll
  for (int i = 0; i < 8; ++i) { f[i] = x[i]; f[i + 8] = y[i]; }
  return f;
}

// Async copy 16 bytes global -> LDS, tracked by ASYNCcnt (CDNA5 path).
static __device__ __forceinline__ void async_b128(unsigned lds_addr, const void* gptr) {
  asm volatile("global_load_async_to_lds_b128 %0, %1, off"
               :: "v"(lds_addr), "v"((unsigned long long)(size_t)gptr)
               : "memory");
}
static __device__ __forceinline__ void wait_async0() {
  asm volatile("s_wait_asynccnt 0x0" ::: "memory");
}
static __device__ __forceinline__ unsigned lds_addr_of(const void* p) {
  // Generic->LDS mapping is truncation to the low 32 bits (ISA 10.2 aperture table).
  return (unsigned)(size_t)p;
}

static __device__ __forceinline__ float rmax16(float x) {
#pragma unroll
  for (int m = 8; m >= 1; m >>= 1) x = fmaxf(x, __shfl_xor(x, m, 16));
  return x;
}
static __device__ __forceinline__ float rsum16(float x) {
#pragma unroll
  for (int m = 8; m >= 1; m >>= 1) x += __shfl_xor(x, m, 16);
  return x;
}

// ---------------------------------------------------------------------------
// QKV projection: Out[b,h,s,hd] = scale * sum_k X[m,k] * W[n,k]  (y = x @ W.T)
// f32 inputs converted to bf16 while staging into LDS.
// Block: 128 threads (4 waves). Tile: 128(M) x 64(N), BK=32; each wave owns
// a 32x64 sub-tile -> 8 WMMAs per K-step with 4 B-fragments reused twice.
// ---------------------------------------------------------------------------
__global__ __launch_bounds__(128)
void gemm_qkv_kernel(const float* __restrict__ X, const float* __restrict__ W,
                     bf16_t* __restrict__ Out, float scale) {
  __shared__ __align__(16) bf16_t sA[128 * 40];
  __shared__ __align__(16) bf16_t sB[64 * 40];
  const int tid  = threadIdx.x;
  const int lane = tid & 31, wave = tid >> 5;
  const int half = lane >> 4, ln = lane & 15, lo = half * 8;
  const int m0 = blockIdx.y * 128, n0 = blockIdx.x * 64;
  const int rB = tid >> 1, cB = (tid & 1) * 16;

  v8f acc[2][4];
#pragma unroll
  for (int mi = 0; mi < 2; ++mi)
#pragma unroll
    for (int t = 0; t < 4; ++t) acc[mi][t] = vzero8();

  const float* gA = X + (size_t)(m0 + tid) * D_;          // one row per thread
  const float* gB = W + (size_t)(n0 + rB) * D_ + cB;

  for (int k0 = 0; k0 < D_; k0 += 32) {
    const float4* a4 = (const float4*)(gA + k0);
#pragma unroll
    for (int j = 0; j < 8; ++j) {
      float4 q = a4[j];
      sA[tid * 40 + 4 * j + 0] = (bf16_t)q.x;
      sA[tid * 40 + 4 * j + 1] = (bf16_t)q.y;
      sA[tid * 40 + 4 * j + 2] = (bf16_t)q.z;
      sA[tid * 40 + 4 * j + 3] = (bf16_t)q.w;
    }
    const float4* b4 = (const float4*)(gB + k0);
#pragma unroll
    for (int j = 0; j < 4; ++j) {
      float4 q = b4[j];
      sB[rB * 40 + cB + 4 * j + 0] = (bf16_t)q.x;
      sB[rB * 40 + cB + 4 * j + 1] = (bf16_t)q.y;
      sB[rB * 40 + cB + 4 * j + 2] = (bf16_t)q.z;
      sB[rB * 40 + cB + 4 * j + 3] = (bf16_t)q.w;
    }
    if (k0 + 32 < D_) {
      __builtin_prefetch(gA + k0 + 32, 0, 1);
      __builtin_prefetch(gB + k0 + 32, 0, 1);
    }
    __syncthreads();

    v16bf a0 = load_frag(sA + (wave * 32 + ln) * 40, lo);
    v16bf a1 = load_frag(sA + (wave * 32 + 16 + ln) * 40, lo);
#pragma unroll
    for (int t = 0; t < 4; ++t) {
      v16bf b = load_frag(sB + (t * 16 + ln) * 40, lo);
      acc[0][t] = __builtin_amdgcn_wmma_f32_16x16x32_bf16(
          false, a0, false, b, (short)0, acc[0][t], false, false);
      acc[1][t] = __builtin_amdgcn_wmma_f32_16x16x32_bf16(
          false, a1, false, b, (short)0, acc[1][t], false, false);
    }
    __syncthreads();
  }

  // Store head-split bf16: [B, NH, S, HD]
#pragma unroll
  for (int mi = 0; mi < 2; ++mi)
#pragma unroll
    for (int t = 0; t < 4; ++t) {
      const int n  = n0 + t * 16 + ln;
      const int h  = n >> 6, hd = n & 63;
#pragma unroll
      for (int rr = 0; rr < 8; ++rr) {
        const int m  = m0 + wave * 32 + mi * 16 + rr + half * 8;
        const int bb = m >> 11, ss = m & (S_ - 1);
        Out[(((size_t)bb * NH_ + h) * S_ + ss) * HD_ + hd] =
            (bf16_t)(acc[mi][t][rr] * scale);
      }
    }
}

// ---------------------------------------------------------------------------
// Flash attention: one block per (b*NH, q-tile of 64 rows). 4 waves, each owns
// 16 q rows. K tile staged via async-to-LDS; V staged with register transpose.
// ---------------------------------------------------------------------------
__global__ __launch_bounds__(128)
void attn_kernel(const bf16_t* __restrict__ Q, const bf16_t* __restrict__ Kc,
                 const bf16_t* __restrict__ Vc, const int* __restrict__ mask,
                 bf16_t* __restrict__ Ctx) {
  __shared__ __align__(16) bf16_t sK[64 * 72];    // [key][hd]
  __shared__ __align__(16) bf16_t sVT[64 * 72];   // [hd][key] (transposed)
  __shared__ __align__(16) bf16_t sP[4][16 * 72]; // per-wave P tile [qrow][key]

  const int tid  = threadIdx.x, lane = tid & 31, wave = tid >> 5;
  const int half = lane >> 4, ln = lane & 15, lo = half * 8;
  const int bh = blockIdx.y, b = bh >> 4, h = bh & 15;
  const int q0 = blockIdx.x * 64;
  const size_t base = (size_t)bh * S_ * HD_;

  // Q A-fragments (16 rows x 64 HD): loaded once, directly from global.
  const bf16_t* Qp = Q + base + (size_t)(q0 + wave * 16 + ln) * HD_;
  v16bf aq0 = load_frag(Qp, lo);
  v16bf aq1 = load_frag(Qp + 32, lo);

  v8f O[4];
#pragma unroll
  for (int t = 0; t < 4; ++t) O[t] = vzero8();
  float mrun[8], lrun[8];
#pragma unroll
  for (int i = 0; i < 8; ++i) { mrun[i] = -3.0e38f; lrun[i] = 0.0f; }

  const int* mk = mask + b * S_;
  const int rr = tid >> 1;            // key row 0..63 for cooperative staging
  const int c0 = (tid & 1) * 32;      // hd half
  const bf16_t* gK = Kc + base + (size_t)rr * HD_ + c0;
  const bf16_t* gV = Vc + base + (size_t)rr * HD_ + c0;
  const unsigned ldsK = lds_addr_of(sK + rr * 72 + c0);

  for (int kb = 0; kb < S_; kb += 64) {
    const size_t off = (size_t)kb * HD_;
    // Stage K tile row-major with async-to-LDS (ASYNCcnt-tracked, no VGPR trip).
#pragma unroll
    for (int j = 0; j < 4; ++j)
      async_b128(ldsK + 16u * j, gK + off + 8 * j);
    // Stage V tile transposed: sVT[hd][key] (needs register transpose).
    v8bf v0 = *(const v8bf*)(gV + off);
    v8bf v1 = *(const v8bf*)(gV + off + 8);
    v8bf v2 = *(const v8bf*)(gV + off + 16);
    v8bf v3 = *(const v8bf*)(gV + off + 24);
#pragma unroll
    for (int j = 0; j < 8; ++j) {
      sVT[(c0 + j) * 72 + rr]      = v0[j];
      sVT[(c0 + 8 + j) * 72 + rr]  = v1[j];
      sVT[(c0 + 16 + j) * 72 + rr] = v2[j];
      sVT[(c0 + 24 + j) * 72 + rr] = v3[j];
    }
    wait_async0();          // barrier waits DScnt only; flush ASYNCcnt ourselves
    __syncthreads();

    // Scores: S = Q (16x64) * K^T, two WMMAs per 16x16 tile.
    v8f sc[4];
#pragma unroll
    for (int t = 0; t < 4; ++t) {
      v16bf bk0 = load_frag(sK + (t * 16 + ln) * 72, lo);
      v16bf bk1 = load_frag(sK + (t * 16 + ln) * 72 + 32, lo);
      v8f cts = vzero8();
      cts = __builtin_amdgcn_wmma_f32_16x16x32_bf16(false, aq0, false, bk0, (short)0, cts, false, false);
      cts = __builtin_amdgcn_wmma_f32_16x16x32_bf16(false, aq1, false, bk1, (short)0, cts, false, false);
      sc[t] = cts;
    }

    // Mask (column = key index): mask==0 -> -inf.
#pragma unroll
    for (int t = 0; t < 4; ++t) {
      if (mk[kb + t * 16 + ln] == 0) {
#pragma unroll
        for (int i = 0; i < 8; ++i) sc[t][i] = -3.0e38f;
      }
    }

    // Online softmax. C-layout: component i is row (i + 8*half), cols across 16 lanes.
    v8f mx = sc[0];
#pragma unroll
    for (int t = 1; t < 4; ++t)
#pragma unroll
      for (int i = 0; i < 8; ++i) mx[i] = fmaxf(mx[i], sc[t][i]);
    float alpha[8];
#pragma unroll
    for (int i = 0; i < 8; ++i) {
      float rm = rmax16(mx[i]);
      float nm = fmaxf(mrun[i], rm);
      alpha[i] = __expf(mrun[i] - nm);
      mrun[i] = nm;
    }
    float rows[8];
#pragma unroll
    for (int i = 0; i < 8; ++i) rows[i] = 0.0f;
#pragma unroll
    for (int t = 0; t < 4; ++t)
#pragma unroll
      for (int i = 0; i < 8; ++i) {
        float p = __expf(sc[t][i] - mrun[i]);
        sc[t][i] = p;
        rows[i] += p;
      }
#pragma unroll
    for (int i = 0; i < 8; ++i) lrun[i] = lrun[i] * alpha[i] + rsum16(rows[i]);
#pragma unroll
    for (int t = 0; t < 4; ++t)
#pragma unroll
      for (int i = 0; i < 8; ++i) O[t][i] *= alpha[i];

    // P round-trip through LDS: C-layout -> A-fragment layout.
    bf16_t* pw = &sP[wave][0];
#pragma unroll
    for (int t = 0; t < 4; ++t)
#pragma unroll
      for (int i = 0; i < 8; ++i)
        pw[(i + half * 8) * 72 + t * 16 + ln] = (bf16_t)sc[t][i];
    __syncthreads();

    v16bf p0 = load_frag(pw + ln * 72, lo);       // keys 0..31
    v16bf p1 = load_frag(pw + ln * 72 + 32, lo);  // keys 32..63
#pragma unroll
    for (int t = 0; t < 4; ++t) {
      v16bf bv0 = load_frag(sVT + (t * 16 + ln) * 72, lo);
      v16bf bv1 = load_frag(sVT + (t * 16 + ln) * 72 + 32, lo);
      O[t] = __builtin_amdgcn_wmma_f32_16x16x32_bf16(false, p0, false, bv0, (short)0, O[t], false, false);
      O[t] = __builtin_amdgcn_wmma_f32_16x16x32_bf16(false, p1, false, bv1, (short)0, O[t], false, false);
    }
    __syncthreads();
  }

  // Normalize and store ctx (heads re-concatenated): [B, S, D] bf16.
#pragma unroll
  for (int t = 0; t < 4; ++t) {
    const int hd = t * 16 + ln;
#pragma unroll
    for (int i = 0; i < 8; ++i) {
      const int s = q0 + wave * 16 + i + half * 8;
      float denom = lrun[i];
      float val = denom > 0.0f ? O[t][i] / denom : 0.0f;
      Ctx[((size_t)b * S_ + s) * D_ + h * HD_ + hd] = (bf16_t)val;
    }
  }
}

// ---------------------------------------------------------------------------
// Output projection + bias + exact-erf GELU: out = gelu(ctx @ W_o.T + b_o), f32.
// bf16 A staged via async-to-LDS; f32 W converted through registers.
// Tile 128(M) x 64(N), BK=32; 8 WMMAs per wave per K-step.
// ---------------------------------------------------------------------------
__global__ __launch_bounds__(128)
void gemm_out_kernel(const bf16_t* __restrict__ Ctx, const float* __restrict__ W,
                     const float* __restrict__ bias, float* __restrict__ Out) {
  __shared__ __align__(16) bf16_t sA[128 * 40];
  __shared__ __align__(16) bf16_t sB[64 * 40];
  const int tid  = threadIdx.x;
  const int lane = tid & 31, wave = tid >> 5;
  const int half = lane >> 4, ln = lane & 15, lo = half * 8;
  const int m0 = blockIdx.y * 128, n0 = blockIdx.x * 64;
  const int rB = tid >> 1, cB = (tid & 1) * 16;

  v8f acc[2][4];
#pragma unroll
  for (int mi = 0; mi < 2; ++mi)
#pragma unroll
    for (int t = 0; t < 4; ++t) acc[mi][t] = vzero8();

  const bf16_t* gA = Ctx + (size_t)(m0 + tid) * D_;       // one row per thread
  const float*  gB = W + (size_t)(n0 + rB) * D_ + cB;
  const unsigned ldsA = lds_addr_of(sA + tid * 40);

  for (int k0 = 0; k0 < D_; k0 += 32) {
    // Async-copy this thread's 64B row slab of bf16 ctx into LDS.
#pragma unroll
    for (int j = 0; j < 4; ++j)
      async_b128(ldsA + 16u * j, gA + k0 + 8 * j);
    const float4* b4 = (const float4*)(gB + k0);
#pragma unroll
    for (int j = 0; j < 4; ++j) {
      float4 q = b4[j];
      sB[rB * 40 + cB + 4 * j + 0] = (bf16_t)q.x;
      sB[rB * 40 + cB + 4 * j + 1] = (bf16_t)q.y;
      sB[rB * 40 + cB + 4 * j + 2] = (bf16_t)q.z;
      sB[rB * 40 + cB + 4 * j + 3] = (bf16_t)q.w;
    }
    if (k0 + 32 < D_) {
      __builtin_prefetch(gA + k0 + 32, 0, 1);
      __builtin_prefetch(gB + k0 + 32, 0, 1);
    }
    wait_async0();
    __syncthreads();

    v16bf a0 = load_frag(sA + (wave * 32 + ln) * 40, lo);
    v16bf a1 = load_frag(sA + (wave * 32 + 16 + ln) * 40, lo);
#pragma unroll
    for (int t = 0; t < 4; ++t) {
      v16bf b = load_frag(sB + (t * 16 + ln) * 40, lo);
      acc[0][t] = __builtin_amdgcn_wmma_f32_16x16x32_bf16(
          false, a0, false, b, (short)0, acc[0][t], false, false);
      acc[1][t] = __builtin_amdgcn_wmma_f32_16x16x32_bf16(
          false, a1, false, b, (short)0, acc[1][t], false, false);
    }
    __syncthreads();
  }

#pragma unroll
  for (int mi = 0; mi < 2; ++mi)
#pragma unroll
    for (int t = 0; t < 4; ++t) {
      const int n = n0 + t * 16 + ln;
      const float bz = bias[n];
#pragma unroll
      for (int rr = 0; rr < 8; ++rr) {
        const int m = m0 + wave * 32 + mi * 16 + rr + half * 8;
        float x = acc[mi][t][rr] + bz;
        float g = 0.5f * x * (1.0f + erff(x * 0.70710678118654752f));
        Out[(size_t)m * D_ + n] = g;
      }
    }
}

// ---------------------------------------------------------------------------
extern "C" void kernel_launch(void* const* d_in, const int* in_sizes, int n_in,
                              void* d_out, int out_size, void* d_ws, size_t ws_size,
                              hipStream_t stream) {
  const float* value = (const float*)d_in[0];
  const float* key_t = (const float*)d_in[1];
  const float* query = (const float*)d_in[2];
  const int*   mask  = (const int*)d_in[3];
  const float* W_q   = (const float*)d_in[4];
  const float* W_k   = (const float*)d_in[5];
  const float* W_v   = (const float*)d_in[6];
  const float* W_o   = (const float*)d_in[7];
  const float* b_o   = (const float*)d_in[8];
  float* out = (float*)d_out;

  const size_t nElem = (size_t)B_ * S_ * D_;  // 8,388,608
  bf16_t* Qb = (bf16_t*)d_ws;                 // ws usage: 4 * nElem * 2 = 64 MB
  bf16_t* Kb = Qb + nElem;
  bf16_t* Vb = Kb + nElem;
  bf16_t* Cb = Vb + nElem;

  dim3 blk(128);
  dim3 gGemm(D_ / 64, (B_ * S_) / 128);  // 16 x 64
  gemm_qkv_kernel<<<gGemm, blk, 0, stream>>>(query, W_q, Qb, 0.125f);  // 1/sqrt(HD)
  gemm_qkv_kernel<<<gGemm, blk, 0, stream>>>(key_t, W_k, Kb, 1.0f);
  gemm_qkv_kernel<<<gGemm, blk, 0, stream>>>(value, W_v, Vb, 1.0f);

  dim3 gAttn(S_ / 64, B_ * NH_);         // 32 x 64
  attn_kernel<<<gAttn, blk, 0, stream>>>(Qb, Kb, Vb, mask, Cb);

  gemm_out_kernel<<<gGemm, blk, 0, stream>>>(Cb, W_o, b_o, out);
}
